// MultiHeadAttention_5720896438843
// MI455X (gfx1250) — matrix operand order repaired and checked by reading, not compiled, not run on hardware
//
#include <hip/hip_runtime.h>
#include <hip/hip_bf16.h>

#define MODEL 1024
#define HEADS 16
#define DH    64
#define SEQ   2048
#define BATCH 2

typedef __attribute__((ext_vector_type(16))) __bf16 bf16x16;
typedef __attribute__((ext_vector_type(8)))  __bf16 bf16x8;
typedef __attribute__((ext_vector_type(8)))  float  f32x8;
typedef __attribute__((ext_vector_type(4)))  unsigned uint32x4;
typedef __attribute__((ext_vector_type(8)))  int      int32x8;
typedef __attribute__((ext_vector_type(4)))  int      int32x4;

// Build a 16-half WMMA fragment from two aligned 16-byte segments.
__device__ inline bf16x16 frag2(const __bf16* p0, const __bf16* p1) {
  bf16x8 lo = *(const bf16x8*)p0;
  bf16x8 hi = *(const bf16x8*)p1;
  bf16x16 r;
#pragma unroll
  for (int i = 0; i < 8; ++i) { r[i] = lo[i]; r[i + 8] = hi[i]; }
  return r;
}

__device__ inline f32x8 wmma_bf16(bf16x16 a, bf16x16 b, f32x8 c) {
  return __builtin_amdgcn_wmma_f32_16x16x32_bf16(false, a, false, b, (short)0, c,
                                                 false, false);
}

// ---- Tensor Data Mover: 2-D bf16 tile (global -> LDS), D# per ISA ch.8 ------
// tensor_d0/stride0 in elements; tile is tile0 x tile1 rows, packed row-major
// into LDS at lds_addr. data_size=1 (2 bytes). Groups 2/3 zero (2-D tile).
// This toolchain exposes the 6-arg builtin (4 descriptor groups + pad + cpol).
__device__ inline void tdm_load_2d(unsigned lds_addr, const void* gptr,
                                   unsigned tensor_d0, unsigned tensor_d1,
                                   unsigned stride0, unsigned tile0,
                                   unsigned tile1) {
  unsigned long long ga = (unsigned long long)gptr;
  uint32x4 g0;
  g0[0] = 1u;                                               // count=1 (valid D#)
  g0[1] = lds_addr;                                         // bits 63:32
  g0[2] = (unsigned)(ga & 0xFFFFFFFFu);                     // addr[31:0]
  g0[3] = (unsigned)((ga >> 32) & 0x01FFFFFFu) | (2u << 30);// addr[56:32], type=2
  int32x8 g1;
  g1[0] = (int)(1u << 16);                                  // data_size=1 (2B)
  g1[1] = (int)((tensor_d0 & 0xFFFFu) << 16);               // tensor_dim0[15:0]
  g1[2] = (int)((tensor_d0 >> 16) | ((tensor_d1 & 0xFFFFu) << 16));
  g1[3] = (int)((tensor_d1 >> 16) | (tile0 << 16));         // tile_dim0
  g1[4] = (int)(tile1 & 0xFFFFu);                           // tile_dim1 (dim2=0)
  g1[5] = (int)stride0;                                     // dim0_stride[31:0]
  g1[6] = 0;                                                // stride hi / dim1_stride
  g1[7] = 0;
  int32x4 z4 = {0, 0, 0, 0};
  int32x8 z8 = {0, 0, 0, 0, 0, 0, 0, 0};
  __builtin_amdgcn_tensor_load_to_lds(g0, g1, z4, z4, z8, 0);
}

// ---- GEMM staging helpers (convert-to-bf16 into LDS, 32 elems/thread) -------
__device__ inline void stage_a32(const float* s, __bf16* d) {
  const float4* s4 = (const float4*)s;
#pragma unroll
  for (int i = 0; i < 8; ++i) {
    float4 v = s4[i];
    d[i * 4 + 0] = (__bf16)v.x; d[i * 4 + 1] = (__bf16)v.y;
    d[i * 4 + 2] = (__bf16)v.z; d[i * 4 + 3] = (__bf16)v.w;
  }
}
__device__ inline void stage_a32(const __bf16* s, __bf16* d) {
  const uint4* s4 = (const uint4*)s;
  uint4* d4 = (uint4*)d;
#pragma unroll
  for (int i = 0; i < 4; ++i) d4[i] = s4[i];
}

template <typename TIN>
__device__ inline void stage_tiles(const TIN* __restrict__ A,
                                   const float* __restrict__ W, int m0, int n0,
                                   int k0, __bf16* As, __bf16* Bs, int tid) {
  {  // 128x64 activation tile, row-major pitch 72
    int r = tid >> 1, seg = tid & 1;
    stage_a32(A + (size_t)(m0 + r) * MODEL + k0 + seg * 32, As + r * 72 + seg * 32);
  }
  {  // 64x128 weight tile, stored transposed Bs[n][k] pitch 72
    int kk = tid >> 2, seg = tid & 3;
    const float4* s4 = (const float4*)(W + (size_t)(k0 + kk) * MODEL + n0 + seg * 32);
#pragma unroll
    for (int j = 0; j < 8; ++j) {
      float4 v = s4[j];
      int n = seg * 32 + j * 4;
      Bs[(n + 0) * 72 + kk] = (__bf16)v.x; Bs[(n + 1) * 72 + kk] = (__bf16)v.y;
      Bs[(n + 2) * 72 + kk] = (__bf16)v.z; Bs[(n + 3) * 72 + kk] = (__bf16)v.w;
    }
  }
}

// MODE 0: f32 out [M x 1024] + bias (final projection to d_out)
// MODE 1: bf16 out [b,h,n,64], scaled 1/sqrt(64) (Q)
// MODE 2: bf16 out [b,h,n,64] (K)
// MODE 3: bf16 out [b,h,64,n] transposed (V)
template <int MODE, typename TIN>
__global__ __launch_bounds__(256) void gemm_wmma(const TIN* __restrict__ A,
                                                 const float* __restrict__ W,
                                                 const float* __restrict__ bias,
                                                 void* __restrict__ outp) {
  __shared__ __attribute__((aligned(16))) __bf16 As[2][128 * 72];
  __shared__ __attribute__((aligned(16))) __bf16 Bs[2][128 * 72];
  const int m0 = (blockIdx.x >> 3) * 128;
  const int n0 = (blockIdx.x & 7) * 128;
  const int tid = threadIdx.x;
  const int wave = tid >> 5, lane = tid & 31, lr = lane & 15, hs = lane >> 4;
  const int wm = wave >> 1, wn = wave & 1;  // 4x2 wave grid -> 32x64 per wave

  f32x8 acc[2][4];
#pragma unroll
  for (int i = 0; i < 2; ++i)
#pragma unroll
    for (int j = 0; j < 4; ++j)
#pragma unroll
      for (int e = 0; e < 8; ++e) acc[i][j][e] = 0.0f;

  stage_tiles(A, W, m0, n0, 0, As[0], Bs[0], tid);
  int cur = 0;
  for (int k0 = 0; k0 < MODEL; k0 += 64) {
    __syncthreads();  // staged tile 'cur' visible; buffer cur^1 free to write
    if (k0 + 64 < MODEL)
      stage_tiles(A, W, m0, n0, k0 + 64, As[cur ^ 1], Bs[cur ^ 1], tid);
#pragma unroll
    for (int kk = 0; kk < 2; ++kk) {
      bf16x16 af[2], bfm[4];
#pragma unroll
      for (int am = 0; am < 2; ++am) {
        const __bf16* p = As[cur] + (wm * 32 + am * 16 + lr) * 72 + kk * 32 + hs * 8;
        af[am] = frag2(p, p + 16);
      }
#pragma unroll
      for (int bn = 0; bn < 4; ++bn) {
        const __bf16* p = Bs[cur] + (wn * 64 + bn * 16 + lr) * 72 + kk * 32 + hs * 8;
        bfm[bn] = frag2(p, p + 16);
      }
#pragma unroll
      for (int am = 0; am < 2; ++am)
#pragma unroll
        for (int bn = 0; bn < 4; ++bn)
          acc[am][bn] = wmma_bf16(af[am], bfm[bn], acc[am][bn]);
    }
    cur ^= 1;
  }

  // Epilogue: C/D layout VGPR r -> row (r + hs*8), col = lane%16
#pragma unroll
  for (int am = 0; am < 2; ++am)
#pragma unroll
    for (int bn = 0; bn < 4; ++bn) {
      int gmb = m0 + wm * 32 + am * 16;
      int gc  = n0 + wn * 64 + bn * 16 + lr;
      float bv = bias[gc];
#pragma unroll
      for (int r = 0; r < 8; ++r) {
        int gm = gmb + r + hs * 8;
        float val = acc[am][bn][r] + bv;
        if constexpr (MODE == 0) {
          ((float*)outp)[(size_t)gm * MODEL + gc] = val;
        } else {
          int b = gm >> 11, n = gm & (SEQ - 1);
          int hh = gc >> 6, d = gc & (DH - 1);
          __bf16* out = (__bf16*)outp;
          if constexpr (MODE == 1) val *= 0.125f;  // 1/sqrt(head_size)
          if constexpr (MODE == 3)
            out[(((size_t)(b * HEADS + hh) * DH + d) * SEQ) + n] = (__bf16)val;
          else
            out[(((size_t)(b * HEADS + hh) * SEQ + n) * DH) + d] = (__bf16)val;
        }
      }
    }
}

// ---- Flash-attention: K/V tiles DMA'd by the TDM into double-buffered LDS ---
__global__ __launch_bounds__(256) void attn_wmma(const __bf16* __restrict__ qS,
                                                 const __bf16* __restrict__ kS,
                                                 const __bf16* __restrict__ vT,
                                                 __bf16* __restrict__ ctx) {
  __shared__ __attribute__((aligned(16))) __bf16 Kt[2][64][64];    // 16 KB
  __shared__ __attribute__((aligned(16))) __bf16 Vt[2][64][64];    // 16 KB
  __shared__ __attribute__((aligned(16))) __bf16 Plds[8 * 32 * 64];// 32 KB
  const int tid = threadIdx.x, wave = tid >> 5, lane = tid & 31;
  const int lr = lane & 15, hs = lane >> 4;
  const int bh = blockIdx.x >> 3;
  const int b = bh >> 4, h = bh & 15;
  const int q0 = (blockIdx.x & 7) * 256 + wave * 32;
  const __bf16* Q = qS + (size_t)bh * SEQ * DH;
  const __bf16* K = kS + (size_t)bh * SEQ * DH;
  const __bf16* V = vT + (size_t)bh * DH * SEQ;
  __bf16* P = Plds + wave * (32 * 64);

  // Q fragments (held for the whole pass): A-layout, dh split into two K=32 chunks
  bf16x16 aq[2][2];
#pragma unroll
  for (int m = 0; m < 2; ++m) {
    int row = q0 + m * 16 + lr;
#pragma unroll
    for (int kk = 0; kk < 2; ++kk) {
      const __bf16* p = Q + (size_t)row * DH + kk * 32 + hs * 8;
      aq[m][kk] = frag2(p, p + 16);
    }
  }

  f32x8 c[2][4];
  float mrow[2][8], lrow[2][8];
#pragma unroll
  for (int m = 0; m < 2; ++m)
#pragma unroll
    for (int r = 0; r < 8; ++r) { mrow[m][r] = -3.0e38f; lrow[m][r] = 0.0f; }
#pragma unroll
  for (int m = 0; m < 2; ++m)
#pragma unroll
    for (int d = 0; d < 4; ++d)
#pragma unroll
      for (int e = 0; e < 8; ++e) c[m][d][e] = 0.0f;

  // Preload key chunk 0: wave0 issues two TDM descriptors (K rows, V^T rows).
  if (tid < 32) {
    tdm_load_2d((unsigned)(uintptr_t)&Kt[0][0][0], K, DH, SEQ, DH, 64, 64);
    tdm_load_2d((unsigned)(uintptr_t)&Vt[0][0][0], V, SEQ, DH, SEQ, 64, 64);
  }

  int cur = 0;
  for (int kc = 0; kc < SEQ; kc += 64) {
    if (tid < 32) {
      if (kc + 64 < SEQ) {  // kick off next chunk, then wait for current pair
        tdm_load_2d((unsigned)(uintptr_t)&Kt[cur ^ 1][0][0],
                    K + (size_t)(kc + 64) * DH, DH, SEQ, DH, 64, 64);
        tdm_load_2d((unsigned)(uintptr_t)&Vt[cur ^ 1][0][0],
                    V + (size_t)(kc + 64), SEQ, DH, SEQ, 64, 64);
        __builtin_amdgcn_s_wait_tensorcnt(2);  // in-order: current pair done
      } else {
        __builtin_amdgcn_s_wait_tensorcnt(0);
      }
    }
    __syncthreads();  // K/V tile 'cur' ready for all waves

    // K fragments from LDS: B-layout, col = key row, contiguous K runs
    bf16x16 bk[4][2];
#pragma unroll
    for (int nb = 0; nb < 4; ++nb)
#pragma unroll
      for (int kk = 0; kk < 2; ++kk) {
        const __bf16* p = &Kt[cur][nb * 16 + lr][kk * 32 + hs * 8];
        bk[nb][kk] = frag2(p, p + 16);
      }
    // scores S = Q K^T (pre-scaled via Q)
    f32x8 s[2][4];
#pragma unroll
    for (int m = 0; m < 2; ++m)
#pragma unroll
      for (int nb = 0; nb < 4; ++nb) {
        f32x8 z;
#pragma unroll
        for (int e = 0; e < 8; ++e) z[e] = 0.0f;
        z = wmma_bf16(aq[m][0], bk[nb][0], z);
        z = wmma_bf16(aq[m][1], bk[nb][1], z);
        s[m][nb] = z;
      }
    // online softmax; row of VGPR r lives in 16-lane half-groups
#pragma unroll
    for (int m = 0; m < 2; ++m)
#pragma unroll
      for (int r = 0; r < 8; ++r) {
        float mx = fmaxf(fmaxf(s[m][0][r], s[m][1][r]), fmaxf(s[m][2][r], s[m][3][r]));
#pragma unroll
        for (int off = 1; off < 16; off <<= 1) mx = fmaxf(mx, __shfl_xor(mx, off, 32));
        float mold = mrow[m][r];
        float mn = fmaxf(mold, mx);
        float sc = __expf(mold - mn);
        mrow[m][r] = mn;
        float rs = 0.0f;
        int prow = m * 16 + r + hs * 8;
#pragma unroll
        for (int nb = 0; nb < 4; ++nb) {
          float pv = __expf(s[m][nb][r] - mn);
          rs += pv;
          P[prow * 64 + nb * 16 + lr] = (__bf16)pv;  // stage in D-layout order
        }
#pragma unroll
        for (int off = 1; off < 16; off <<= 1) rs += __shfl_xor(rs, off, 32);
        lrow[m][r] = lrow[m][r] * sc + rs;
#pragma unroll
        for (int db = 0; db < 4; ++db) c[m][db][r] *= sc;
      }
    __asm__ volatile("" ::: "memory");  // wave-private LDS: DS in-order per wave
    // ctx += P @ V ; P as A-fragments from LDS, V^T tile gives contiguous B-frags
#pragma unroll
    for (int kk2 = 0; kk2 < 2; ++kk2) {
      bf16x16 pa[2];
#pragma unroll
      for (int m = 0; m < 2; ++m) {
        const __bf16* p = P + (m * 16 + lr) * 64 + kk2 * 32 + hs * 8;
        pa[m] = frag2(p, p + 16);
      }
      bf16x16 vb[4];
#pragma unroll
      for (int db = 0; db < 4; ++db) {
        const __bf16* p = &Vt[cur][db * 16 + lr][kk2 * 32 + hs * 8];
        vb[db] = frag2(p, p + 16);
      }
#pragma unroll
      for (int m = 0; m < 2; ++m)
#pragma unroll
        for (int db = 0; db < 4; ++db)
          c[m][db] = wmma_bf16(pa[m], vb[db], c[m][db]);
    }
    __syncthreads();  // all waves done with tile 'cur' before TDM reuses it
    cur ^= 1;
  }

  // finalize: divide by row sums, write ctx in [B*N, 1024] bf16 for out-proj
#pragma unroll
  for (int m = 0; m < 2; ++m)
#pragma unroll
    for (int r = 0; r < 8; ++r) {
      float inv = 1.0f / lrow[m][r];
      int row = q0 + m * 16 + r + hs * 8;
#pragma unroll
      for (int db = 0; db < 4; ++db)
        ctx[((size_t)(b * SEQ + row)) * MODEL + h * DH + db * 16 + lr] =
            (__bf16)(c[m][db][r] * inv);
    }
}

extern "C" void kernel_launch(void* const* d_in, const int* in_sizes, int n_in,
                              void* d_out, int out_size, void* d_ws, size_t ws_size,
                              hipStream_t stream) {
  const float* query = (const float*)d_in[0];
  const float* key   = (const float*)d_in[1];
  const float* value = (const float*)d_in[2];
  const float* Wq = (const float*)d_in[3]; const float* bq = (const float*)d_in[4];
  const float* Wk = (const float*)d_in[5]; const float* bk = (const float*)d_in[6];
  const float* Wv = (const float*)d_in[7]; const float* bv = (const float*)d_in[8];
  const float* Wo = (const float*)d_in[9]; const float* bo = (const float*)d_in[10];

  const size_t per = (size_t)BATCH * HEADS * SEQ * DH;  // 4M elems = 8 MiB bf16
  __bf16* qS  = (__bf16*)d_ws;          // [b,h,n,64], pre-scaled by 1/8
  __bf16* kS  = qS + per;               // [b,h,n,64]
  __bf16* vT  = kS + per;               // [b,h,64,n]
  __bf16* ctx = vT + per;               // [b*n, 1024]

  dim3 blk(256);
  dim3 gg(256);  // (4096/128) * (1024/128)
  gemm_wmma<1, float><<<gg, blk, 0, stream>>>(query, Wq, bq, (void*)qS);
  gemm_wmma<2, float><<<gg, blk, 0, stream>>>(key,   Wk, bk, (void*)kS);
  gemm_wmma<3, float><<<gg, blk, 0, stream>>>(value, Wv, bv, (void*)vT);
  attn_wmma<<<dim3(256), blk, 0, stream>>>(qS, kS, vT, ctx);
  gemm_wmma<0, __bf16><<<gg, blk, 0, stream>>>(ctx, Wo, bo, d_out);
}
